// MSA_20667382628451
// MI455X (gfx1250) — matrix operand order repaired and checked by reading, not compiled, not run on hardware
//
#include <hip/hip_runtime.h>

#define MDIM 32768   // B*S
#define NDIM 1024    // D
#define KDIM 1024    // D

typedef __attribute__((ext_vector_type(16))) __bf16 v16bf;
typedef __attribute__((ext_vector_type(8)))  __bf16 v8bf;
typedef __attribute__((ext_vector_type(4)))  __bf16 v4bf;
typedef __attribute__((ext_vector_type(8)))  float  v8f;
typedef __attribute__((ext_vector_type(4)))  float  f32x4;
typedef __attribute__((ext_vector_type(2)))  float  f32x2;

__device__ __forceinline__ v16bf concat8(v8bf a, v8bf b) {
    return __builtin_shufflevector(a, b, 0,1,2,3,4,5,6,7,8,9,10,11,12,13,14,15);
}

// ---------------------------------------------------------------------------
// One-shot split: f32 -> bf16 hi + bf16 lo correction (bf16x3 decomposition).
// ---------------------------------------------------------------------------
__global__ __launch_bounds__(256) void split_f32_to_bf16x2(
    const float* __restrict__ src, __bf16* __restrict__ hi,
    __bf16* __restrict__ lo, int n4)
{
    const int i = blockIdx.x * blockDim.x + threadIdx.x;
    if (i >= n4) return;
    f32x4 v = ((const f32x4*)src)[i];
    v4bf h, l;
#pragma unroll
    for (int e = 0; e < 4; ++e) {
        __bf16 hh = (__bf16)v[e];
        h[e] = hh;
        l[e] = (__bf16)(v[e] - (float)hh);
    }
    ((v4bf*)hi)[i] = h;
    ((v4bf*)lo)[i] = l;
}

// ---------------------------------------------------------------------------
// Y[m,n] = sum_k A[m,k]*W[n,k] + bias[n], A and W pre-split into bf16 hi/lo.
// bf16x3: Ah*Wh + Ah*Wl + Al*Wh, fp32 WMMA accumulate. Zero conversion VALU
// in the K-loop. Wave tile 32x64 (2 A-frags x 4 B-frags). Block 8 waves,
// tile 128x128. SPLIT_OUT: emit split bf16 planes (intermediates) or fp32.
// ---------------------------------------------------------------------------
template <bool SPLIT_OUT>
__global__ __launch_bounds__(256) void gemm_bf16x3(
    const __bf16* __restrict__ Ah, const __bf16* __restrict__ Al,
    const __bf16* __restrict__ Wh, const __bf16* __restrict__ Wl,
    const float*  __restrict__ bias,
    float* __restrict__ Yf, __bf16* __restrict__ Yh, __bf16* __restrict__ Yl)
{
    const int lane = threadIdx.x & 31;
    const int wave = threadIdx.x >> 5;
    const int half = lane >> 4;   // 0: lanes 0-15, 1: lanes 16-31
    const int lr   = lane & 15;

    const int m0 = blockIdx.x * 128 + (wave & 3) * 32;
    const int n0 = blockIdx.y * 128 + (wave >> 2) * 64;

    v8f acc[2][4] = {};

    const size_t arow0 = (size_t)(m0 + lr) * KDIM;
    const size_t arow1 = arow0 + (size_t)16 * KDIM;

    for (int kb = 0; kb < KDIM; kb += 32) {
        // A fragment (16x32 bf16): lane half 0 -> K 0..15,
        // half 1 -> K 16..31; two b128 loads per fragment.
        const int ka = kb + half * 16;
        const __bf16* p0h = Ah + arow0 + ka;
        const __bf16* p1h = Ah + arow1 + ka;
        const __bf16* p0l = Al + arow0 + ka;
        const __bf16* p1l = Al + arow1 + ka;
        v16bf ah[2], al[2];
        ah[0] = concat8(*(const v8bf*)p0h, *(const v8bf*)(p0h + 8));
        ah[1] = concat8(*(const v8bf*)p1h, *(const v8bf*)(p1h + 8));
        al[0] = concat8(*(const v8bf*)p0l, *(const v8bf*)(p0l + 8));
        al[1] = concat8(*(const v8bf*)p1l, *(const v8bf*)(p1l + 8));

#pragma unroll
        for (int j = 0; j < 4; ++j) {
            // B fragment (32x16 bf16, same contiguous K order as A): lane half selects K 0-15/16-31,
            // 16 contiguous bf16 -> one 32B (2x b128) load per plane.
            const size_t wrow = (size_t)(n0 + j * 16 + lr) * KDIM + kb + half * 16;
            v16bf bh = *(const v16bf*)(Wh + wrow);
            v16bf bl = *(const v16bf*)(Wl + wrow);
#pragma unroll
            for (int r = 0; r < 2; ++r) {
                acc[r][j] = __builtin_amdgcn_wmma_f32_16x16x32_bf16(
                    false, ah[r], false, bh, (short)0, acc[r][j], false, false);
                acc[r][j] = __builtin_amdgcn_wmma_f32_16x16x32_bf16(
                    false, ah[r], false, bl, (short)0, acc[r][j], false, false);
                acc[r][j] = __builtin_amdgcn_wmma_f32_16x16x32_bf16(
                    false, al[r], false, bh, (short)0, acc[r][j], false, false);
            }
        }
    }

    // Epilogue: C/D layout — VGPR i: row = base + half*8 + i, lane -> column.
#pragma unroll
    for (int j = 0; j < 4; ++j) {
        const int ncol = n0 + j * 16 + lr;
        const float bv = bias[ncol];
#pragma unroll
        for (int r = 0; r < 2; ++r) {
#pragma unroll
            for (int i = 0; i < 8; ++i) {
                const int row = m0 + r * 16 + half * 8 + i;
                const float val = acc[r][j][i] + bv;
                const size_t o = (size_t)row * NDIM + ncol;
                if (SPLIT_OUT) {
                    __bf16 h = (__bf16)val;
                    Yh[o] = h;
                    Yl[o] = (__bf16)(val - (float)h);
                } else {
                    Yf[o] = val;
                }
            }
        }
    }
}

// ---------------------------------------------------------------------------
// Per-token head-axis attention (scores over heads, softmax, ctx). Inputs and
// outputs are split bf16 hi/lo planes; math in fp32. One wave per token.
// Vectorized LDS access: q-row hoisted to registers, k read as b128, v as b64.
// ---------------------------------------------------------------------------
__global__ __launch_bounds__(32) void attn_headaxis_kernel(
    const __bf16* __restrict__ QH, const __bf16* __restrict__ QL,
    const __bf16* __restrict__ KH, const __bf16* __restrict__ KL,
    const __bf16* __restrict__ VH, const __bf16* __restrict__ VL,
    __bf16* __restrict__ CH, __bf16* __restrict__ CL)
{
    __shared__ float sq[16 * 68];  // padded rows to dodge bank conflicts
    __shared__ float sk[1024];
    __shared__ float sv[1024];
    __shared__ float sp[256];

    const int lane = threadIdx.x;
    const size_t tok = (size_t)blockIdx.x * 1024;

    for (int i4 = lane; i4 < 256; i4 += 32) {  // 4 elems per iter
        const int idx = i4 * 4;
        v4bf qh4 = ((const v4bf*)(QH + tok))[i4];
        v4bf ql4 = ((const v4bf*)(QL + tok))[i4];
        v4bf kh4 = ((const v4bf*)(KH + tok))[i4];
        v4bf kl4 = ((const v4bf*)(KL + tok))[i4];
        v4bf vh4 = ((const v4bf*)(VH + tok))[i4];
        v4bf vl4 = ((const v4bf*)(VL + tok))[i4];
        f32x4 q4, k4, v4;
#pragma unroll
        for (int e = 0; e < 4; ++e) {
            q4[e] = (float)qh4[e] + (float)ql4[e];
            k4[e] = (float)kh4[e] + (float)kl4[e];
            v4[e] = (float)vh4[e] + (float)vl4[e];
        }
        const int h = idx >> 6, d = idx & 63;
        *(f32x4*)&sq[h * 68 + d] = q4;   // 68%4==0 and d%4==0 -> 16B aligned
        *(f32x4*)&sk[idx] = k4;
        *(f32x4*)&sv[idx] = v4;
    }
    __syncthreads();

    if (lane < 16) {  // lane = head h, owns the full 16-wide score row
        f32x4 qreg[16];  // hoist the 64-wide q row into VGPRs once
#pragma unroll
        for (int c = 0; c < 16; ++c) qreg[c] = *(const f32x4*)&sq[lane * 68 + c * 4];

        float sc[16];
        float mx = -3.402823466e38f;
#pragma unroll
        for (int g = 0; g < 16; ++g) {
            float s = 0.0f;
#pragma unroll
            for (int c = 0; c < 16; ++c) {
                f32x4 kk = *(const f32x4*)&sk[g * 64 + c * 4];  // ds_load_b128
                s += qreg[c][0] * kk[0] + qreg[c][1] * kk[1]
                   + qreg[c][2] * kk[2] + qreg[c][3] * kk[3];
            }
            s *= 0.125f;  // 1/sqrt(64)
            sc[g] = s;
            mx = fmaxf(mx, s);
        }
        float sum = 0.0f;
#pragma unroll
        for (int g = 0; g < 16; ++g) { float e = __expf(sc[g] - mx); sc[g] = e; sum += e; }
        const float inv = 1.0f / sum;
#pragma unroll
        for (int g = 0; g < 16; ++g) sp[lane * 16 + g] = sc[g] * inv;
    }
    __syncthreads();

    const int d0 = lane * 2;  // each lane owns 2 of the 64 head-dims
    for (int h = 0; h < 16; ++h) {
        f32x4 pr[4];  // hoist the 16-wide prob row (broadcast reads)
#pragma unroll
        for (int c = 0; c < 4; ++c) pr[c] = *(const f32x4*)&sp[h * 16 + c * 4];
        float c0 = 0.0f, c1 = 0.0f;
#pragma unroll
        for (int g = 0; g < 16; ++g) {
            const float p = pr[g >> 2][g & 3];
            f32x2 vv = *(const f32x2*)&sv[g * 64 + d0];  // ds_load_b64
            c0 += p * vv[0];
            c1 += p * vv[1];
        }
        const size_t o = tok + h * 64 + d0;
        __bf16 h0 = (__bf16)c0, h1 = (__bf16)c1;
        CH[o]     = h0;  CL[o]     = (__bf16)(c0 - (float)h0);
        CH[o + 1] = h1;  CL[o + 1] = (__bf16)(c1 - (float)h1);
    }
}

extern "C" void kernel_launch(void* const* d_in, const int* in_sizes, int n_in,
                              void* d_out, int out_size, void* d_ws, size_t ws_size,
                              hipStream_t stream) {
    const float* x  = (const float*)d_in[0];
    const float* wq = (const float*)d_in[1];
    const float* bq = (const float*)d_in[2];
    const float* wk = (const float*)d_in[3];
    const float* bk = (const float*)d_in[4];
    const float* wv = (const float*)d_in[5];
    const float* bv = (const float*)d_in[6];
    const float* wo = (const float*)d_in[7];
    const float* bo = (const float*)d_in[8];
    float* out = (float*)d_out;

    const size_t MD = (size_t)MDIM * KDIM;  // 33.5M elems
    const size_t WD = (size_t)NDIM * KDIM;  // 1M elems

    __bf16* p = (__bf16*)d_ws;              // ~553 MB of bf16 planes
    __bf16 *xh = p,        *xl = xh + MD;
    __bf16 *qh = xl + MD,  *ql = qh + MD;
    __bf16 *kh = ql + MD,  *kl = kh + MD;
    __bf16 *vh = kl + MD,  *vl = vh + MD;
    __bf16 *wqh = vl + MD,  *wql = wqh + WD;
    __bf16 *wkh = wql + WD, *wkl = wkh + WD;
    __bf16 *wvh = wkl + WD, *wvl = wvh + WD;
    __bf16 *woh = wvl + WD, *wol = woh + WD;

    // 1) split x and weights into bf16 hi/lo planes (one cheap pass)
    split_f32_to_bf16x2<<<(int)(MD / 4 / 256), 256, 0, stream>>>(x,  xh,  xl,  (int)(MD / 4));
    split_f32_to_bf16x2<<<(int)(WD / 4 / 256), 256, 0, stream>>>(wq, wqh, wql, (int)(WD / 4));
    split_f32_to_bf16x2<<<(int)(WD / 4 / 256), 256, 0, stream>>>(wk, wkh, wkl, (int)(WD / 4));
    split_f32_to_bf16x2<<<(int)(WD / 4 / 256), 256, 0, stream>>>(wv, wvh, wvl, (int)(WD / 4));
    split_f32_to_bf16x2<<<(int)(WD / 4 / 256), 256, 0, stream>>>(wo, woh, wol, (int)(WD / 4));

    // 2) Q/K/V projections (split-in, split-out)
    dim3 grid(MDIM / 128, NDIM / 128);  // (256, 8)
    gemm_bf16x3<true><<<grid, 256, 0, stream>>>(xh, xl, wqh, wql, bq, nullptr, qh, ql);
    gemm_bf16x3<true><<<grid, 256, 0, stream>>>(xh, xl, wkh, wkl, bk, nullptr, kh, kl);
    gemm_bf16x3<true><<<grid, 256, 0, stream>>>(xh, xl, wvh, wvl, bv, nullptr, vh, vl);

    // 3) per-token attention; ctx overwrites the q planes (token staged in LDS)
    attn_headaxis_kernel<<<MDIM, 32, 0, stream>>>(qh, ql, kh, kl, vh, vl, qh, ql);

    // 4) output projection -> fp32 d_out
    gemm_bf16x3<false><<<grid, 256, 0, stream>>>(qh, ql, woh, wol, bo, out, nullptr, nullptr);
}
